// LinearizedMoeExperts_6751688589474
// MI455X (gfx1250) — compile-verified, matrix operand order checked by reading, not run on hardware
//
#include <hip/hip_runtime.h>

// ---------------- problem constants ----------------
#define NEXP   8
#define HID    2048
#define INTER  1408
#define TOK    2048
#define TOPK   2
#define SLOTS  (TOK * TOPK)   // 4096
#define MAXS   4096           // per-expert list capacity (worst case)

#define CHUNK  128            // K-chunk staged in LDS
#define LSTR   (CHUNK + 8)    // padded row stride (bf16 elems) = 272B -> bank-conflict-free

typedef __attribute__((ext_vector_type(16))) __bf16 v16bf;
typedef __attribute__((ext_vector_type(8)))  __bf16 v8bf;
typedef __attribute__((ext_vector_type(8)))  float  v8f;

// ---------------------------------------------------------------------------
// ISA 16-bit A-matrix 16x32 layout (CDNA5 ISA):
//   lanes 0-15  (half=0): M=lane,   elems 0-7 -> K=0..7,  elems 8-15 -> K=16..23
//   lanes 16-31 (half=1): M=lane,   elems 0-7 -> K=8..15, elems 8-15 -> K=24..31
// B (K x N = W^T) fragments use the mirrored per-lane scheme on row-major W,
// giving two contiguous runs per lane -> coalesced b128 loads.
// ---------------------------------------------------------------------------
__device__ __forceinline__ v16bf load_frag_f32(const float* __restrict__ row,
                                               int k0, int half) {
  const float4* pa = reinterpret_cast<const float4*>(row + k0 + half * 8);
  const float4* pb = reinterpret_cast<const float4*>(row + k0 + 16 + half * 8);
  float4 a0 = pa[0], a1 = pa[1];
  float4 b0 = pb[0], b1 = pb[1];
  v16bf v;
  v[0]  = (__bf16)a0.x; v[1]  = (__bf16)a0.y; v[2]  = (__bf16)a0.z; v[3]  = (__bf16)a0.w;
  v[4]  = (__bf16)a1.x; v[5]  = (__bf16)a1.y; v[6]  = (__bf16)a1.z; v[7]  = (__bf16)a1.w;
  v[8]  = (__bf16)b0.x; v[9]  = (__bf16)b0.y; v[10] = (__bf16)b0.z; v[11] = (__bf16)b0.w;
  v[12] = (__bf16)b1.x; v[13] = (__bf16)b1.y; v[14] = (__bf16)b1.z; v[15] = (__bf16)b1.w;
  return v;
}

// A fragment from LDS-staged bf16 tile (ds_load_b128 x2 per lane)
__device__ __forceinline__ v16bf lds_frag(const __bf16* base, int row,
                                          int kk, int half) {
  const __bf16* p = base + row * LSTR + kk;
  v8bf a = *reinterpret_cast<const v8bf*>(p + half * 8);
  v8bf b = *reinterpret_cast<const v8bf*>(p + 16 + half * 8);
  v16bf v;
#pragma unroll
  for (int j = 0; j < 8; ++j) { v[j] = a[j]; v[8 + j] = b[j]; }
  return v;
}

// ---------------------------------------------------------------------------
// Kernel 1: routing. slot id = token*2 + k. Per-expert compacted lists in ws.
// ---------------------------------------------------------------------------
__global__ void moe_route(const int* __restrict__ idx,
                          const float* __restrict__ wts,
                          int* __restrict__ counts,
                          int* __restrict__ lists,
                          float* __restrict__ wl) {
  int s = blockIdx.x * blockDim.x + threadIdx.x;
  if (s >= SLOTS) return;
  int e = idx[s];
  int pos = atomicAdd(&counts[e], 1);
  lists[e * MAXS + pos] = s;
  wl[e * MAXS + pos] = wts[s];
}

// ---------------------------------------------------------------------------
// Kernel 2: grouped gate/up GEMM + SiLU.
// Block = 256 threads = 8 wave32s covering a 32(M) x 128(N) tile.
// A (32 gathered activation rows) is staged per-K-chunk into LDS as bf16 once
// per block; each wave owns a 16-column strip and all 32 rows:
// per K-step: 4 ds_load_b128 (A) + 8 global_load_b128 (B) -> 4 WMMAs.
// ---------------------------------------------------------------------------
__global__ __launch_bounds__(256)
void moe_gate_up(const float* __restrict__ hs,
                 const float* __restrict__ Wg,
                 const float* __restrict__ Wu,
                 const int* __restrict__ counts,
                 const int* __restrict__ lists,
                 __bf16* __restrict__ hws) {
  __shared__ __bf16 As[32 * LSTR];

  const int e  = blockIdx.z;
  const int ne = counts[e];
  const int m0 = blockIdx.y * 32;
  if (m0 >= ne) return;                       // uniform: EXEC stays all-ones
  const int n0 = blockIdx.x * 128;

  const int wave = threadIdx.x >> 5;          // 8 N-strips of 16 columns
  const int lane = threadIdx.x & 31;
  const int half = lane >> 4;
  const int l15  = lane & 15;

  const int ncol = n0 + wave * 16 + l15;
  const float* __restrict__ grow = Wg + ((size_t)e * INTER + ncol) * HID;
  const float* __restrict__ urow = Wu + ((size_t)e * INTER + ncol) * HID;

  // staging assignment: thread -> (row, 16-col slice)
  const int sr = threadIdx.x >> 3;            // 0..31
  const int sc = (threadIdx.x & 7) * 16;      // 0,16,..,112
  const int srm = min(m0 + sr, ne - 1);
  const float* __restrict__ srow =
      hs + (size_t)(lists[e * MAXS + srm] >> 1) * HID;

  v8f accg0 = {0.f,0.f,0.f,0.f,0.f,0.f,0.f,0.f};
  v8f accg1 = accg0, accu0 = accg0, accu1 = accg0;

  for (int kc = 0; kc < HID; kc += CHUNK) {
    __syncthreads();                          // protect As reuse
    // ---- stage A chunk: 16 f32 -> 16 bf16 per thread ----
    const float4* g4 = reinterpret_cast<const float4*>(srow + kc + sc);
    float4 f0 = g4[0], f1 = g4[1], f2 = g4[2], f3 = g4[3];
    v8bf s0, s1;
    s0[0]=(__bf16)f0.x; s0[1]=(__bf16)f0.y; s0[2]=(__bf16)f0.z; s0[3]=(__bf16)f0.w;
    s0[4]=(__bf16)f1.x; s0[5]=(__bf16)f1.y; s0[6]=(__bf16)f1.z; s0[7]=(__bf16)f1.w;
    s1[0]=(__bf16)f2.x; s1[1]=(__bf16)f2.y; s1[2]=(__bf16)f2.z; s1[3]=(__bf16)f2.w;
    s1[4]=(__bf16)f3.x; s1[5]=(__bf16)f3.y; s1[6]=(__bf16)f3.z; s1[7]=(__bf16)f3.w;
    *reinterpret_cast<v8bf*>(&As[sr * LSTR + sc])     = s0;
    *reinterpret_cast<v8bf*>(&As[sr * LSTR + sc + 8]) = s1;
    __syncthreads();

#pragma unroll
    for (int kk = 0; kk < CHUNK; kk += 32) {
      __builtin_prefetch(grow + kc + kk + 256, 0, 1);   // global_prefetch_b8
      __builtin_prefetch(urow + kc + kk + 256, 0, 1);
      v16bf a0 = lds_frag(As, l15,      kk, half);
      v16bf a1 = lds_frag(As, 16 + l15, kk, half);
      v16bf bg = load_frag_f32(grow, kc + kk, half);
      v16bf bu = load_frag_f32(urow, kc + kk, half);
      accg0 = __builtin_amdgcn_wmma_f32_16x16x32_bf16(false, a0, false, bg,
                                                      (short)0, accg0, false, false);
      accg1 = __builtin_amdgcn_wmma_f32_16x16x32_bf16(false, a1, false, bg,
                                                      (short)0, accg1, false, false);
      accu0 = __builtin_amdgcn_wmma_f32_16x16x32_bf16(false, a0, false, bu,
                                                      (short)0, accu0, false, false);
      accu1 = __builtin_amdgcn_wmma_f32_16x16x32_bf16(false, a1, false, bu,
                                                      (short)0, accu1, false, false);
    }
  }

  // C/D layout: VGPR r, lanes 0-15 -> M=r, lanes 16-31 -> M=r+8; N=l15
#pragma unroll
  for (int mi = 0; mi < 2; ++mi) {
    v8f g = mi ? accg1 : accg0;
    v8f u = mi ? accu1 : accu0;
#pragma unroll
    for (int r = 0; r < 8; ++r) {
      int m = m0 + mi * 16 + half * 8 + r;
      if (m < ne) {
        int oslot = lists[e * MAXS + m];
        float gv = g[r], uv = u[r];
        float hval = (gv / (1.f + __expf(-gv))) * uv;   // silu(g)*u
        hws[(size_t)oslot * INTER + ncol] = (__bf16)hval;
      }
    }
  }
}

// ---------------------------------------------------------------------------
// Kernel 3: grouped down-projection + weighted scatter.
// Same 32x128 tiling; A = bf16 intermediate staged to LDS (pure 16B copies).
// Each out element receives exactly TOPK=2 atomic adds (commutative -> bitwise
// deterministic).
// ---------------------------------------------------------------------------
__global__ __launch_bounds__(256)
void moe_down(const float* __restrict__ Wd,
              const __bf16* __restrict__ hws,
              const int* __restrict__ counts,
              const int* __restrict__ lists,
              const float* __restrict__ wl,
              float* __restrict__ out) {
  __shared__ __bf16 As[32 * LSTR];

  const int e  = blockIdx.z;
  const int ne = counts[e];
  const int m0 = blockIdx.y * 32;
  if (m0 >= ne) return;
  const int n0 = blockIdx.x * 128;

  const int wave = threadIdx.x >> 5;
  const int lane = threadIdx.x & 31;
  const int half = lane >> 4;
  const int l15  = lane & 15;

  const int ncol = n0 + wave * 16 + l15;
  const float* __restrict__ brow = Wd + ((size_t)e * HID + ncol) * INTER;

  const int sr = threadIdx.x >> 3;
  const int sc = (threadIdx.x & 7) * 16;
  const int srm = min(m0 + sr, ne - 1);
  const __bf16* __restrict__ srow =
      hws + (size_t)lists[e * MAXS + srm] * INTER;

  v8f acc0 = {0.f,0.f,0.f,0.f,0.f,0.f,0.f,0.f};
  v8f acc1 = acc0;

  for (int kc = 0; kc < INTER; kc += CHUNK) {
    __syncthreads();
    v8bf s0 = *reinterpret_cast<const v8bf*>(srow + kc + sc);
    v8bf s1 = *reinterpret_cast<const v8bf*>(srow + kc + sc + 8);
    *reinterpret_cast<v8bf*>(&As[sr * LSTR + sc])     = s0;
    *reinterpret_cast<v8bf*>(&As[sr * LSTR + sc + 8]) = s1;
    __syncthreads();

#pragma unroll
    for (int kk = 0; kk < CHUNK; kk += 32) {
      __builtin_prefetch(brow + kc + kk + 256, 0, 1);
      v16bf a0 = lds_frag(As, l15,      kk, half);
      v16bf a1 = lds_frag(As, 16 + l15, kk, half);
      v16bf b  = load_frag_f32(brow, kc + kk, half);
      acc0 = __builtin_amdgcn_wmma_f32_16x16x32_bf16(false, a0, false, b,
                                                     (short)0, acc0, false, false);
      acc1 = __builtin_amdgcn_wmma_f32_16x16x32_bf16(false, a1, false, b,
                                                     (short)0, acc1, false, false);
    }
  }

#pragma unroll
  for (int mi = 0; mi < 2; ++mi) {
    v8f a = mi ? acc1 : acc0;
#pragma unroll
    for (int r = 0; r < 8; ++r) {
      int m = m0 + mi * 16 + half * 8 + r;
      if (m < ne) {
        int   s2  = lists[e * MAXS + m];
        float wgt = wl[e * MAXS + m];
        int   tok = s2 >> 1;
        atomicAdd(&out[(size_t)tok * HID + ncol], wgt * a[r]);
      }
    }
  }
}

// ---------------------------------------------------------------------------
// Launcher. ws layout:
//   [0,256)            counts (int[8], zeroed per call)
//   [256, +128KB)      lists  (int[8][4096])
//   [+128KB, +128KB)   wl     (float[8][4096])
//   [.., +11.5MB)      hws    (bf16[4096][1408])
// ---------------------------------------------------------------------------
extern "C" void kernel_launch(void* const* d_in, const int* in_sizes, int n_in,
                              void* d_out, int out_size, void* d_ws, size_t ws_size,
                              hipStream_t stream) {
  (void)in_sizes; (void)n_in; (void)out_size; (void)ws_size;
  const float* hs  = (const float*)d_in[0];
  const int*   idx = (const int*)d_in[1];
  const float* tkw = (const float*)d_in[2];
  const float* Wg  = (const float*)d_in[3];
  const float* Wu  = (const float*)d_in[4];
  const float* Wd  = (const float*)d_in[5];
  float* out = (float*)d_out;

  char* ws = (char*)d_ws;
  int*    counts = (int*)ws;
  int*    lists  = (int*)(ws + 256);
  float*  wl     = (float*)(ws + 256 + NEXP * MAXS * 4);
  __bf16* hws    = (__bf16*)(ws + 256 + 2 * NEXP * MAXS * 4);

  hipMemsetAsync(counts, 0, 256, stream);
  hipMemsetAsync(out, 0, (size_t)TOK * HID * sizeof(float), stream);

  moe_route<<<SLOTS / 256, 256, 0, stream>>>(idx, tkw, counts, lists, wl);

  dim3 g1(INTER / 128, MAXS / 32, NEXP);   // 11 x 128 x 8
  moe_gate_up<<<g1, 256, 0, stream>>>(hs, Wg, Wu, counts, lists, hws);

  dim3 g2(HID / 128, MAXS / 32, NEXP);     // 16 x 128 x 8
  moe_down<<<g2, 256, 0, stream>>>(Wd, hws, counts, lists, wl, out);
}
